// DeltaNet_31877247271581
// MI455X (gfx1250) — matrix operand order repaired and checked by reading, not compiled, not run on hardware
//
#include <hip/hip_runtime.h>
#include <hip/hip_bf16.h>
#include <math.h>
#include <stdint.h>

// ---------------- problem constants (from reference) ----------------
constexpr int Bc = 2, Lc = 4096, Dc = 1024;
constexpr int Hc = 4, DKc = 256, DVc = 256;
constexpr int CONVK = 4;
constexpr int GQP = 128, NSc = 6, GIN = GQP + NSc, GHc = 134;
constexpr int CH = 32;
constexpr int NCHUNK = Lc / CH;
constexpr float FLOORc = 0.02f;
constexpr int DSPLIT = 4;           // dv-split of the delta recurrence (columns of S independent)
constexpr int DVS = DVc / DSPLIT;   // 64 dv columns per block

typedef __bf16 bf16_t;
typedef __attribute__((ext_vector_type(16))) __bf16 v16bf;
typedef __attribute__((ext_vector_type(8)))  __bf16 v8bf;
typedef __attribute__((ext_vector_type(4)))  __bf16 v4bf;
typedef __attribute__((ext_vector_type(8)))  float  v8f;

// ---------------- WMMA fragment helpers (ISA 7.12.2, wave32) ----------
__device__ __forceinline__ v16bf frag_a(const bf16_t* base, int ld, int lane) {
  int m = lane & 15, half = lane >> 4;
  v8bf lo = *(const v8bf*)(base + m * ld + 8 * half);
  v8bf hi = *(const v8bf*)(base + m * ld + 16 + 8 * half);
  return __builtin_shufflevector(lo, hi, 0, 1, 2, 3, 4, 5, 6, 7, 8, 9, 10, 11, 12, 13, 14, 15);
}
__device__ __forceinline__ v16bf frag_bT(const bf16_t* base, int ld, int lane) {
  int n = lane & 15, half = lane >> 4;
  v8bf lo = *(const v8bf*)(base + n * ld + 8 * half);
  v8bf hi = *(const v8bf*)(base + n * ld + 16 + 8 * half);
  return __builtin_shufflevector(lo, hi, 0, 1, 2, 3, 4, 5, 6, 7, 8, 9, 10, 11, 12, 13, 14, 15);
}
__device__ __forceinline__ v8f wmma_bf16(v16bf a, v16bf b, v8f c) {
  return __builtin_amdgcn_wmma_f32_16x16x32_bf16(false, a, false, b, (short)0, c, false, false);
}
// software-pipelined 8-step K accumulation (K=256)
__device__ __forceinline__ v8f mm_k256(const bf16_t* Abase, int lda, const bf16_t* Bbase, int ldb,
                                       int lane, v8f acc) {
  v16bf af = frag_a(Abase, lda, lane);
  v16bf bfr = frag_bT(Bbase, ldb, lane);
#pragma unroll
  for (int kk = 0; kk < 8; ++kk) {
    v16bf afn = af, bfn = bfr;
    if (kk < 7) {
      afn = frag_a(Abase + (kk + 1) * 32, lda, lane);
      bfn = frag_bT(Bbase + (kk + 1) * 32, ldb, lane);
    }
    acc = wmma_bf16(af, bfr, acc);
    af = afn; bfr = bfn;
  }
  return acc;
}

// ---------------- Tensor Data Mover (TDM) support ----------------
#if __has_builtin(__builtin_amdgcn_tensor_load_to_lds)
#define HAVE_TDM 1
typedef __attribute__((ext_vector_type(4))) unsigned int v4u_t;
typedef __attribute__((ext_vector_type(8))) int v8i_t;
typedef __attribute__((ext_vector_type(4))) int v4i_t;

__device__ __forceinline__ void tdm_load_tile_f32(const float* gsrc, unsigned lds_byte_addr,
                                                  int rows, int cols, int row_stride_elems) {
  unsigned long long ga = (unsigned long long)(uintptr_t)gsrc;
  v4u_t g0;
  g0[0] = 1u;                                               // count=1, user descriptor
  g0[1] = lds_byte_addr;                                    // lds_addr
  g0[2] = (unsigned)(ga & 0xffffffffu);                     // global_addr[31:0]
  g0[3] = (unsigned)((ga >> 32) & 0x1ffffffu) | (2u << 30); // global_addr[56:32] | type=2
  unsigned td0 = (unsigned)cols, td1 = (unsigned)rows, s0 = (unsigned)row_stride_elems;
  v8i_t g1;
  g1[0] = (int)(2u << 16);                                  // data_size = 4 bytes
  g1[1] = (int)((td0 & 0xffffu) << 16);
  g1[2] = (int)((td0 >> 16) | ((td1 & 0xffffu) << 16));
  g1[3] = (int)((td1 >> 16) | ((unsigned)(cols & 0xffff) << 16));
  g1[4] = (int)(rows & 0xffff);
  g1[5] = (int)s0;
  g1[6] = 0;
  g1[7] = 0;
  v4i_t z4 = {0, 0, 0, 0};
#if __clang_major__ >= 23
  v8i_t z8 = {0, 0, 0, 0, 0, 0, 0, 0};
  __builtin_amdgcn_tensor_load_to_lds(g0, g1, z4, z4, z8, 0);
#else
  __builtin_amdgcn_tensor_load_to_lds(g0, g1, z4, z4, 0);
#endif
}
#endif

// stage 32 x cols f32 tile into LDS: TDM DMA (wave 0) or cooperative fallback
__device__ __forceinline__ void load_stage(const float* gsrc, float* stage, unsigned stage_addr,
                                           int tid, int wave, int cols) {
#if defined(HAVE_TDM)
  (void)tid;
  if (wave == 0) {
    tdm_load_tile_f32(gsrc, stage_addr, 32, cols, Hc * DKc);
#if __has_builtin(__builtin_amdgcn_s_wait_tensorcnt)
    __builtin_amdgcn_s_wait_tensorcnt((short)0);
#endif
  }
#else
  (void)stage_addr; (void)wave;
  int cf4 = cols >> 2;
  for (int i = tid; i < 32 * cf4; i += 512) {
    int r = i / cf4, c4 = (i % cf4) * 4;
    *(float4*)(stage + r * cols + c4) = *(const float4*)(gsrc + (size_t)r * (Hc * DKc) + c4);
  }
#endif
}

// ---------------- generic GEMM: C[M,N] = A[M,K] @ W[K,N] (+bias) ----------------
template <bool HAS_BIAS>
__global__ void gemm_bf16_kernel(const float* __restrict__ A, const float* __restrict__ W,
                                 const float* __restrict__ bias, float* __restrict__ C,
                                 int M, int N, int K) {
  __shared__ bf16_t As[32 * 32];
  __shared__ bf16_t Bst[64 * 32];
  __shared__ float  Cs[32 * 64];
  int tid = threadIdx.x, lane = tid & 31, wave = tid >> 5;
  int m0 = blockIdx.y * 32, n0 = blockIdx.x * 64;
  int mt = wave & 1, nt = wave >> 1;
  v8f acc = {0.f, 0.f, 0.f, 0.f, 0.f, 0.f, 0.f, 0.f};

  int ar = tid >> 3, ac4 = (tid & 7) * 4;
  int br = tid >> 3, bc8 = (tid & 7) * 8;

  for (int k0 = 0; k0 < K; k0 += 32) {
    if (tid == 0 && k0 + 64 < K) {
      __builtin_prefetch(&A[(size_t)m0 * K + k0 + 64], 0, 0);
      __builtin_prefetch(&W[(size_t)(k0 + 64) * N + n0], 0, 0);
    }
    {
      float4 av = *(const float4*)(A + (size_t)(m0 + ar) * K + k0 + ac4);
      v4bf a4 = {(bf16_t)av.x, (bf16_t)av.y, (bf16_t)av.z, (bf16_t)av.w};
      *(v4bf*)(As + ar * 32 + ac4) = a4;
    }
    {
      const float* wp = W + (size_t)(k0 + br) * N + n0 + bc8;
      float4 w0 = *(const float4*)(wp);
      float4 w1 = *(const float4*)(wp + 4);
      Bst[(bc8 + 0) * 32 + br] = (bf16_t)w0.x;
      Bst[(bc8 + 1) * 32 + br] = (bf16_t)w0.y;
      Bst[(bc8 + 2) * 32 + br] = (bf16_t)w0.z;
      Bst[(bc8 + 3) * 32 + br] = (bf16_t)w0.w;
      Bst[(bc8 + 4) * 32 + br] = (bf16_t)w1.x;
      Bst[(bc8 + 5) * 32 + br] = (bf16_t)w1.y;
      Bst[(bc8 + 6) * 32 + br] = (bf16_t)w1.z;
      Bst[(bc8 + 7) * 32 + br] = (bf16_t)w1.w;
    }
    __syncthreads();
    v16bf af = frag_a(As + mt * 16 * 32, 32, lane);
    v16bf bfr = frag_bT(Bst + nt * 16 * 32, 32, lane);
    acc = wmma_bf16(af, bfr, acc);
    __syncthreads();
  }
  {
    int n = lane & 15, mh = (lane >> 4) * 8;
#pragma unroll
    for (int r = 0; r < 8; ++r) Cs[(mt * 16 + mh + r) * 64 + nt * 16 + n] = acc[r];
  }
  __syncthreads();
  {
    int row = tid >> 3, c8 = (tid & 7) * 8;
    float o[8];
#pragma unroll
    for (int j = 0; j < 8; ++j) {
      o[j] = Cs[row * 64 + c8 + j];
      if (HAS_BIAS) o[j] += bias[n0 + c8 + j];
    }
    float* cp = C + (size_t)(m0 + row) * N + n0 + c8;
    *(float4*)(cp) = make_float4(o[0], o[1], o[2], o[3]);
    *(float4*)(cp + 4) = make_float4(o[4], o[5], o[6], o[7]);
  }
}

// ---------------- depthwise causal conv (K=4) + SiLU ----------------
__global__ void conv_silu_kernel(const float* __restrict__ X, const float* __restrict__ Wc,
                                 float* __restrict__ Y, int C_) {
  size_t idx = (size_t)blockIdx.x * blockDim.x + threadIdx.x;
  size_t total = (size_t)Bc * Lc * C_;
  if (idx >= total) return;
  int c = (int)(idx % C_);
  size_t bl = idx / C_;
  int l = (int)(bl % Lc);
  size_t b = bl / Lc;
  float acc = 0.f;
#pragma unroll
  for (int t = 0; t < CONVK; ++t) {
    int ll = l - (CONVK - 1) + t;
    if (ll >= 0) acc += X[(b * Lc + ll) * C_ + c] * Wc[c * CONVK + t];
  }
  Y[idx] = acc / (1.f + expf(-acc));
}

// ---------------- beta = sigmoid(x @ Wb) ----------------
__global__ void beta_kernel(const float* __restrict__ X, const float* __restrict__ Wb,
                            float* __restrict__ beta) {
  int t = blockIdx.x * blockDim.x + threadIdx.x;
  if (t >= Bc * Lc * Hc) return;
  int h = t % Hc;
  size_t bl = t / Hc;
  const float* xr = X + bl * Dc;
  float s = 0.f;
  for (int i = 0; i < Dc; ++i) s += xr[i] * Wb[i * Hc + h];
  beta[t] = 1.f / (1.f + expf(-s));
}

// ---------------- chunkwise delta rule, dv-split: 1 block per (b,h,dv-slice) ------
// beta folded into inverse columns: inv2[i,j] = inv[i,j]*beta[j]; u = inv2@v, w = inv2@k.
// LDS ~156 KB/block -> 2 workgroups per 320 KB WGP.
__global__ void delta_kernel(const float* __restrict__ qg_, const float* __restrict__ kg_,
                             const float* __restrict__ vg_, const float* __restrict__ betag,
                             float* __restrict__ outg) {
  extern __shared__ char smem[];
  bf16_t* S_t   = (bf16_t*)smem;             // [DVS*256] state slice, S_t[dv][dk]
  bf16_t* qs    = S_t + DVS * 256;           // [32*256] row-major
  bf16_t* ks    = qs + 32 * 256;             // [32*256] row-major
  bf16_t* ks_t  = ks + 32 * 256;             // [256*32]  A of k^T@u_t; B of inv2@k
  bf16_t* wsb   = ks_t + 256 * 32;           // [32*256] row-major (A of w@S)
  bf16_t* vs_t  = wsb + 32 * 256;            // [DVS*32]
  bf16_t* us_t  = vs_t + DVS * 32;           // [DVS*32]
  bf16_t* ut_t  = us_t + DVS * 32;           // [DVS*32]
  bf16_t* attns = ut_t + DVS * 32;           // [32*32]
  bf16_t* invb  = attns + 32 * 32;           // [32*32]  inv2 (bf16)
  float*  invf  = (float*)(invb + 32 * 32);  // [32*32] f32
  float*  os    = invf + 32 * 32;            // [32*DVS] f32 (q@S partial)
  float*  beta_s = os + 32 * DVS;            // [32] f32
  float*  stage = beta_s + 32;               // [32*256] f32 TDM staging

  int blk = blockIdx.x;
  int sp = blk & (DSPLIT - 1);
  int bh = blk / DSPLIT;
  int b = bh / Hc, h = bh % Hc;
  int dv0 = sp * DVS;
  int tid = threadIdx.x, lane = tid & 31, wave = tid >> 5;  // 512 thr = 16 waves
  unsigned stage_addr = (unsigned)(uintptr_t)(void*)stage;

  for (int i = tid; i < DVS * 256; i += 512) S_t[i] = (bf16_t)0.f;
  __syncthreads();

  for (int c = 0; c < NCHUNK; ++c) {
    int l0 = c * CH;
    size_t cbase = ((size_t)(b * Lc + l0)) * (Hc * DKc) + (size_t)h * DKc;

    if (tid < 32) beta_s[tid] = betag[(size_t)(b * Lc + l0 + tid) * Hc + h];

    // ---- Phase A-q: TDM -> stage; l2norm -> qs ----
    load_stage(qg_ + cbase, stage, stage_addr, tid, wave, 256);
    __syncthreads();
    for (int r = wave; r < CH; r += 16) {
      int d0 = lane * 8;
      const float* sp_ = stage + r * 256 + d0;
      float4 a0 = *(const float4*)sp_, a1 = *(const float4*)(sp_ + 4);
      float qf[8] = {a0.x, a0.y, a0.z, a0.w, a1.x, a1.y, a1.z, a1.w};
      float sq = 0.f;
#pragma unroll
      for (int j = 0; j < 8; ++j) sq += qf[j] * qf[j];
#pragma unroll
      for (int off = 16; off; off >>= 1) sq += __shfl_xor(sq, off, 32);
      float rq = rsqrtf(sq + 1e-6f);
      v8bf qv;
#pragma unroll
      for (int j = 0; j < 8; ++j) qv[j] = (bf16_t)(qf[j] * rq);
      *(v8bf*)(qs + r * 256 + d0) = qv;
    }
    __syncthreads();

    // ---- Phase A-k: TDM -> stage; l2norm -> ks, ks_t (no beta) ----
    load_stage(kg_ + cbase, stage, stage_addr, tid, wave, 256);
    __syncthreads();
    for (int r = wave; r < CH; r += 16) {
      int d0 = lane * 8;
      const float* sp_ = stage + r * 256 + d0;
      float4 a0 = *(const float4*)sp_, a1 = *(const float4*)(sp_ + 4);
      float kf[8] = {a0.x, a0.y, a0.z, a0.w, a1.x, a1.y, a1.z, a1.w};
      float sk = 0.f;
#pragma unroll
      for (int j = 0; j < 8; ++j) sk += kf[j] * kf[j];
#pragma unroll
      for (int off = 16; off; off >>= 1) sk += __shfl_xor(sk, off, 32);
      float rk = rsqrtf(sk + 1e-6f);
      v8bf kv;
#pragma unroll
      for (int j = 0; j < 8; ++j) {
        kv[j] = (bf16_t)(kf[j] * rk);
        ks_t[(d0 + j) * 32 + r] = kv[j];
      }
      *(v8bf*)(ks + r * 256 + d0) = kv;
    }
    __syncthreads();

    // ---- Phase A-v: TDM 32xDVS slice -> stage; -> vs_t (unscaled) ----
    load_stage(vg_ + cbase + dv0, stage, stage_addr, tid, wave, DVS);
    __syncthreads();
    for (int r = wave; r < CH; r += 16) {
      int d0 = lane * 2;
      vs_t[(d0 + 0) * 32 + r] = (bf16_t)stage[r * DVS + d0 + 0];
      vs_t[(d0 + 1) * 32 + r] = (bf16_t)stage[r * DVS + d0 + 1];
    }
    __syncthreads();

    // ---- Phase B: kkT -> inv_init (rows scaled by beta); attn = tril(q@k^T) ----
    if (wave < 8) {
      int which = wave >> 2;               // 0: k@k^T, 1: q@k^T
      int t = wave & 3, mt = t >> 1, nt = t & 1;
      const bf16_t* Ab = (which ? qs : ks) + mt * 16 * 256;
      v8f acc = {0.f, 0.f, 0.f, 0.f, 0.f, 0.f, 0.f, 0.f};
      acc = mm_k256(Ab, 256, ks + nt * 16 * 256, 256, lane, acc);
      int n = lane & 15, mh = (lane >> 4) * 8;
#pragma unroll
      for (int r = 0; r < 8; ++r) {
        int m = mt * 16 + mh + r, nn = nt * 16 + n;
        if (which == 0) invf[m * 32 + nn] = (nn >= m) ? 0.f : -beta_s[m] * acc[r];
        else            attns[m * 32 + nn] = (bf16_t)((nn > m) ? 0.f : acc[r]);
      }
    }
    __syncthreads();

    // ---- Phase B2: forward substitution (wave 0); inv2 = (inv + I) * beta_col ----
    if (wave == 0) {
      volatile float* iv = invf;
      int j = lane;
      for (int i = 1; i < 32; ++i) {
        float upd = 0.f;
        for (int kk = 0; kk < i; ++kk) upd += iv[i * 32 + kk] * iv[kk * 32 + j];
        iv[i * 32 + j] = iv[i * 32 + j] + upd;
        __builtin_amdgcn_wave_barrier();
      }
    }
    __syncthreads();
    for (int idx = tid; idx < 32 * 32; idx += 512) {
      int i = idx >> 5, j = idx & 31;
      invb[idx] = (bf16_t)((invf[idx] + (i == j ? 1.f : 0.f)) * beta_s[j]);
    }
    __syncthreads();

    // ---- Phase C: u = inv2@v (8 tiles -> us_t), w = inv2@k (32 tiles -> wsb) ----
    for (int t = wave; t < 40; t += 16) {
      int n = lane & 15, mh = (lane >> 4) * 8;
      v8f acc = {0.f, 0.f, 0.f, 0.f, 0.f, 0.f, 0.f, 0.f};
      if (t < 8) {
        int mt = t >> 2, nt = t & 3;
        v16bf af = frag_a(invb + mt * 16 * 32, 32, lane);
        v16bf bfr = frag_bT(vs_t + nt * 16 * 32, 32, lane);
        acc = wmma_bf16(af, bfr, acc);
        v8bf uv;
#pragma unroll
        for (int r = 0; r < 8; ++r) uv[r] = (bf16_t)acc[r];
        *(v8bf*)(us_t + (nt * 16 + n) * 32 + mt * 16 + mh) = uv;
      } else {
        int tt = t - 8, mt = tt >> 4, nt = tt & 15;
        v16bf af = frag_a(invb + mt * 16 * 32, 32, lane);
        v16bf bfr = frag_bT(ks_t + nt * 16 * 32, 32, lane);
        acc = wmma_bf16(af, bfr, acc);
#pragma unroll
        for (int r = 0; r < 8; ++r)
          wsb[(mt * 16 + mh + r) * 256 + nt * 16 + n] = (bf16_t)acc[r];
      }
    }
    __syncthreads();

    // ---- Phase D (split): waves 0-7: u_t = u - w@S; waves 8-15: os = q@S ----
    {
      int n = lane & 15, mh = (lane >> 4) * 8;
      if (wave < 8) {
        int mt = wave >> 2, nt = wave & 3;
        v8f acc = {0.f, 0.f, 0.f, 0.f, 0.f, 0.f, 0.f, 0.f};
        acc = mm_k256(wsb + mt * 16 * 256, 256, S_t + nt * 16 * 256, 256, lane, acc);
        v8bf uv = *(const v8bf*)(us_t + (nt * 16 + n) * 32 + mt * 16 + mh);
        v8bf utv;
#pragma unroll
        for (int r = 0; r < 8; ++r) utv[r] = (bf16_t)((float)uv[r] - acc[r]);
        *(v8bf*)(ut_t + (nt * 16 + n) * 32 + mt * 16 + mh) = utv;
      } else {
        int t = wave - 8, mt = t >> 2, nt = t & 3;
        v8f acc = {0.f, 0.f, 0.f, 0.f, 0.f, 0.f, 0.f, 0.f};
        acc = mm_k256(qs + mt * 16 * 256, 256, S_t + nt * 16 * 256, 256, lane, acc);
#pragma unroll
        for (int r = 0; r < 8; ++r) os[(mt * 16 + mh + r) * DVS + nt * 16 + n] = acc[r];
      }
    }
    __syncthreads();

    // ---- Phase D2: o = os + attn@u_t ; store to global ----
    if (wave < 8) {
      int mt = wave >> 2, nt = wave & 3;
      int n = lane & 15, mh = (lane >> 4) * 8;
      v8f acc;
#pragma unroll
      for (int r = 0; r < 8; ++r) acc[r] = os[(mt * 16 + mh + r) * DVS + nt * 16 + n];
      v16bf af = frag_a(attns + mt * 16 * 32, 32, lane);
      v16bf bfr = frag_bT(ut_t + nt * 16 * 32, 32, lane);
      acc = wmma_bf16(af, bfr, acc);
#pragma unroll
      for (int r = 0; r < 8; ++r) {
        int l = l0 + mt * 16 + mh + r;
        outg[((size_t)(b * Lc + l) * Hc + h) * DVc + dv0 + nt * 16 + n] = acc[r];
      }
    }
    __syncthreads();

    // ---- Phase E: S += k^T @ u_t (64 tiles, 4 per wave) ----
#pragma unroll
    for (int ti = 0; ti < 4; ++ti) {
      int idx = wave * 4 + ti;               // 0..63
      int mt = idx >> 2, nt = idx & 3;       // mt: dk tile (16), nt: dv tile (4)
      int n = lane & 15, mh = (lane >> 4) * 8;
      bf16_t* sp_ = S_t + (nt * 16 + n) * 256 + mt * 16 + mh;
      v8bf sv = *(const v8bf*)sp_;
      v8f acc;
#pragma unroll
      for (int r = 0; r < 8; ++r) acc[r] = (float)sv[r];
      v16bf af = frag_a(ks_t + mt * 16 * 32, 32, lane);
      v16bf bfr = frag_bT(ut_t + nt * 16 * 32, 32, lane);
      acc = wmma_bf16(af, bfr, acc);
#pragma unroll
      for (int r = 0; r < 8; ++r) sv[r] = (bf16_t)acc[r];
      *(v8bf*)sp_ = sv;
    }
    __syncthreads();
  }
}

// ---------------- fused FIR streams + gate MLP + softmax + mix + rmsnorm ----------
__global__ void fuse_kernel(const float* __restrict__ vconv, const float* __restrict__ delta,
                            const float* __restrict__ f0, const float* __restrict__ f1,
                            const float* __restrict__ f2, const float* __restrict__ f3,
                            const float* __restrict__ qg, const float* __restrict__ Wf1,
                            const float* __restrict__ bf1, const float* __restrict__ Wf2,
                            const float* __restrict__ bf2, const float* __restrict__ log_temp,
                            const float* __restrict__ onw, float* __restrict__ fused) {
  int blh = blockIdx.x;
  int h = blh % Hc;
  size_t bl = blh / Hc;
  int l = (int)(bl % Lc);
  size_t b = bl / Lc;
  int d = threadIdx.x;
  int cch = h * DVc + d;

  float s[NSc];
  {
    const int KS[4] = {1, 3, 7, 15};
    const float* fw[4] = {f0, f1, f2, f3};
#pragma unroll
    for (int j = 0; j < 4; ++j) {
      float acc = 0.f;
      int Kj = KS[j];
      for (int t = 0; t < Kj; ++t) {
        int ll = l - (Kj - 1) + t;
        if (ll >= 0) acc += vconv[(b * Lc + ll) * (Hc * DVc) + cch] * fw[j][cch * Kj + t];
      }
      s[j] = acc;
    }
  }
  s[4] = delta[((b * Lc + l) * Hc + h) * DVc + d];
  s[5] = vconv[(b * Lc + l) * (Hc * DVc) + cch];

  __shared__ float red[256];
  __shared__ float gate_in[GIN];
  __shared__ float hmid[GHc];
  __shared__ float p_sm[NSc];

  for (int j = 0; j < NSc; ++j) {
    red[d] = s[j];
    __syncthreads();
    for (int off = 128; off; off >>= 1) {
      if (d < off) red[d] += red[d + off];
      __syncthreads();
    }
    if (d == 0) gate_in[GQP + j] = red[0] / (float)DVc;
    __syncthreads();
  }
  if (d < GQP) gate_in[d] = qg[(b * Lc + l) * GQP + d];
  __syncthreads();
  if (d < GHc) {
    float acc = bf1[d];
    for (int i = 0; i < GIN; ++i) acc += gate_in[i] * Wf1[i * GHc + d];
    hmid[d] = 0.5f * acc * (1.f + erff(acc * 0.70710678118654752f));
  }
  __syncthreads();
  if (d < NSc) {
    float acc = bf2[d];
    for (int i = 0; i < GHc; ++i) acc += hmid[i] * Wf2[i * NSc + d];
    p_sm[d] = acc / expf(log_temp[h]);
  }
  __syncthreads();
  if (d == 0) {
    float mx = p_sm[0];
    for (int j = 1; j < NSc; ++j) mx = fmaxf(mx, p_sm[j]);
    float e[NSc], den = 0.f;
    for (int j = 0; j < NSc; ++j) { e[j] = expf(p_sm[j] - mx); den += e[j]; }
    for (int j = 0; j < NSc; ++j) p_sm[j] = e[j] / den * (1.f - NSc * FLOORc) + FLOORc;
  }
  __syncthreads();
  float f = 0.f;
#pragma unroll
  for (int j = 0; j < NSc; ++j) f += p_sm[j] * s[j];
  red[d] = f * f;
  __syncthreads();
  for (int off = 128; off; off >>= 1) {
    if (d < off) red[d] += red[d + off];
    __syncthreads();
  }
  float r = rsqrtf(red[0] / (float)DVc + 1e-5f);
  fused[(b * Lc + l) * (Hc * DVc) + cch] = f * r * onw[d];
}

// ---------------- host launcher ----------------
extern "C" void kernel_launch(void* const* d_in, const int* in_sizes, int n_in,
                              void* d_out, int out_size, void* d_ws, size_t ws_size,
                              hipStream_t stream) {
  (void)in_sizes; (void)n_in; (void)out_size; (void)ws_size;
  const float* x       = (const float*)d_in[0];
  const float* Wq      = (const float*)d_in[1];
  const float* Wk      = (const float*)d_in[2];
  const float* Wv      = (const float*)d_in[3];
  const float* Wb      = (const float*)d_in[4];
  const float* conv_qw = (const float*)d_in[5];
  const float* conv_kw = (const float*)d_in[6];
  const float* conv_vw = (const float*)d_in[7];
  const float* fir0    = (const float*)d_in[8];
  const float* fir1    = (const float*)d_in[9];
  const float* fir2    = (const float*)d_in[10];
  const float* fir3    = (const float*)d_in[11];
  const float* Wqg     = (const float*)d_in[12];
  const float* bqg     = (const float*)d_in[13];
  const float* Wf1     = (const float*)d_in[14];
  const float* bf1     = (const float*)d_in[15];
  const float* Wf2     = (const float*)d_in[16];
  const float* bf2     = (const float*)d_in[17];
  const float* log_temp= (const float*)d_in[18];
  const float* onw     = (const float*)d_in[19];
  const float* Wo      = (const float*)d_in[20];

  const size_t BL = (size_t)Bc * Lc;        // 8192
  float* ws = (float*)d_ws;
  size_t off = 0;
  float* q_lin  = ws + off; off += BL * 1024;
  float* k_lin  = ws + off; off += BL * 1024;
  float* v_lin  = ws + off; off += BL * 1024;
  float* q_conv = ws + off; off += BL * 1024;
  float* k_conv = ws + off; off += BL * 1024;
  float* v_conv = ws + off; off += BL * 1024;
  float* beta   = ws + off; off += BL * Hc;
  float* qg     = ws + off; off += BL * GQP;
  float* dout   = ws + off; off += BL * 1024;
  float* fused  = ws + off; off += BL * 1024;

  gemm_bf16_kernel<false><<<dim3(1024 / 64, (int)(BL / 32)), 256, 0, stream>>>(x, Wq, nullptr, q_lin, (int)BL, 1024, 1024);
  gemm_bf16_kernel<false><<<dim3(1024 / 64, (int)(BL / 32)), 256, 0, stream>>>(x, Wk, nullptr, k_lin, (int)BL, 1024, 1024);
  gemm_bf16_kernel<false><<<dim3(1024 / 64, (int)(BL / 32)), 256, 0, stream>>>(x, Wv, nullptr, v_lin, (int)BL, 1024, 1024);
  gemm_bf16_kernel<true><<<dim3(GQP / 64, (int)(BL / 32)), 256, 0, stream>>>(x, Wqg, bqg, qg, (int)BL, GQP, 1024);

  beta_kernel<<<(int)((BL * Hc + 255) / 256), 256, 0, stream>>>(x, Wb, beta);

  {
    size_t tot = BL * 1024;
    int blocks = (int)((tot + 255) / 256);
    conv_silu_kernel<<<blocks, 256, 0, stream>>>(q_lin, conv_qw, q_conv, 1024);
    conv_silu_kernel<<<blocks, 256, 0, stream>>>(k_lin, conv_kw, k_conv, 1024);
    conv_silu_kernel<<<blocks, 256, 0, stream>>>(v_lin, conv_vw, v_conv, 1024);
  }

  {
    size_t smem = (size_t)(DVS * 256 + 4 * 32 * 256 + 3 * DVS * 32 + 2 * 32 * 32) * sizeof(bf16_t)
                + (size_t)(32 * 32 + 32 * DVS + 32 + 32 * 256) * sizeof(float);  // ~156 KB
    delta_kernel<<<Bc * Hc * DSPLIT, 512, smem, stream>>>(q_conv, k_conv, v_conv, beta, dout);
  }

  fuse_kernel<<<(int)(BL * Hc), 256, 0, stream>>>(v_conv, dout, fir0, fir1, fir2, fir3,
                                                  qg, Wf1, bf1, Wf2, bf2, log_temp, onw, fused);

  gemm_bf16_kernel<false><<<dim3(1024 / 64, (int)(BL / 32)), 256, 0, stream>>>(fused, Wo, nullptr,
                                                                               (float*)d_out, (int)BL, 1024, 1024);
}
